// LSTMLanguageModel_24094766530709
// MI455X (gfx1250) — compile-verified, matrix operand order checked by reading, not run on hardware
//
#include <hip/hip_runtime.h>

typedef __bf16 bf16_t;
typedef __attribute__((ext_vector_type(16))) __bf16 v16bf;
typedef __attribute__((ext_vector_type(8)))  float  v8f;
typedef int v4i_g __attribute__((vector_size(16)));   // GCC-style vector, matches builtin proto

// K index for element e (0..15) of a 16-bit A/B WMMA fragment (ISA 7.12.2):
// VGPR j (=e>>1) holds K = {2j,2j+1} (+8 for lanes 16..31), VGPRs 4..7 are +16.
__device__ __forceinline__ int frag_k(int e, int hi) {
  int vg = e >> 1;
  return ((vg & 4) << 2) | (hi << 3) | ((vg & 3) << 1) | (e & 1);
}

// ---------------- async global -> LDS copy (CDNA5 path), with sync fallback ---
#if defined(__AMDGCN__) && __has_builtin(__builtin_amdgcn_global_load_async_to_lds_b128) && \
    __has_builtin(__builtin_amdgcn_s_wait_asynccnt)
#define USE_ASYNC_LDS 1
#else
#define USE_ASYNC_LDS 0
#endif

__device__ __forceinline__ void copy16_g2l(const void* g, void* l) {
#if USE_ASYNC_LDS
  // (v4i global* src, v4i shared* dst, imm offset, imm cpol)
  __builtin_amdgcn_global_load_async_to_lds_b128(
      (__attribute__((address_space(1))) v4i_g*)(unsigned long long)(__SIZE_TYPE__)g,
      (__attribute__((address_space(3))) v4i_g*)(unsigned)(__SIZE_TYPE__)l,
      0, 0);
#else
  *(v4i_g*)l = *(const v4i_g*)g;
#endif
}

__device__ __forceinline__ void wait_async_all() {
#if USE_ASYNC_LDS
  __builtin_amdgcn_s_wait_asynccnt(0);
#endif
}

// -----------------------------------------------------------------------------
// C[M,N] (f32) = A[M,K] (bf16) * W[N,K]^T (bf16) + bias0 + bias1
// Block: 512 threads = 16 waves (4M x 4N), block tile 128(M) x 256(N),
// wave tile 32x64 (8 accumulators). K staged 32 at a time through
// double-buffered LDS (48 KB) filled with async global->LDS copies.
// -----------------------------------------------------------------------------
__global__ __launch_bounds__(512) void gemm_bf16_wmma(
    const bf16_t* __restrict__ A, const bf16_t* __restrict__ W,
    const float* __restrict__ bias0, const float* __restrict__ bias1,
    float* __restrict__ C, int M, int N, int K) {
  __shared__ bf16_t sA[2][128 * 32];  // 2 x 8 KB
  __shared__ bf16_t sW[2][256 * 32];  // 2 x 16 KB
  const int tid  = threadIdx.x;
  const int lane = tid & 31;
  const int wave = tid >> 5;      // 0..15
  const int wm   = wave & 3;      // 4 waves in M
  const int wn   = wave >> 2;     // 4 waves in N
  const int mblk = blockIdx.x * 128;
  const int nblk = blockIdx.y * 256;

  // Stage one 128x32 A tile + 256x32 W tile (16B chunks, one per lane-slot).
  auto stage = [&](int buf, int k0) {
#pragma unroll
    for (int c = 0; c < 2; ++c) {       // W: 256 rows * 64B = 1024 chunks
      int chunk = tid + c * 512;
      int row = chunk >> 2;
      int off = (chunk & 3) << 4;
      copy16_g2l((const char*)(W + (size_t)(nblk + row) * K + k0) + off,
                 (char*)(&sW[buf][row * 32]) + off);
    }
    {                                    // A: 128 rows * 64B = 512 chunks
      int row = tid >> 2;
      int off = (tid & 3) << 4;
      copy16_g2l((const char*)(A + (size_t)(mblk + row) * K + k0) + off,
                 (char*)(&sA[buf][row * 32]) + off);
    }
  };

  v8f acc[2][4];
#pragma unroll
  for (int mi = 0; mi < 2; ++mi)
#pragma unroll
    for (int j = 0; j < 4; ++j)
#pragma unroll
      for (int r = 0; r < 8; ++r) acc[mi][j][r] = 0.f;

  const int nsteps = K >> 5;
  stage(0, 0);
  wait_async_all();
  __syncthreads();

  const int hi = lane >> 4;
  for (int s = 0; s < nsteps; ++s) {
    const int cur = s & 1;
    if (s + 1 < nsteps) stage(cur ^ 1, (s + 1) << 5);

    v16bf a0, a1;
    {
      const bf16_t* p0 = &sA[cur][(wm * 32 + 0 + (lane & 15)) * 32];
      const bf16_t* p1 = &sA[cur][(wm * 32 + 16 + (lane & 15)) * 32];
#pragma unroll
      for (int e = 0; e < 16; ++e) { a0[e] = p0[frag_k(e, hi)]; a1[e] = p1[frag_k(e, hi)]; }
    }
#pragma unroll
    for (int j = 0; j < 4; ++j) {
      v16bf b;
      const bf16_t* pb = &sW[cur][(wn * 64 + j * 16 + (lane & 15)) * 32];
#pragma unroll
      for (int e = 0; e < 16; ++e) b[e] = pb[frag_k(e, hi)];
      acc[0][j] = __builtin_amdgcn_wmma_f32_16x16x32_bf16(
          false, a0, false, b, (short)0, acc[0][j], false, false);
      acc[1][j] = __builtin_amdgcn_wmma_f32_16x16x32_bf16(
          false, a1, false, b, (short)0, acc[1][j], false, false);
    }

    if (s + 1 < nsteps) wait_async_all();
    __syncthreads();
  }

  const int n = lane & 15;
#pragma unroll
  for (int mi = 0; mi < 2; ++mi) {
#pragma unroll
    for (int j = 0; j < 4; ++j) {
      const int nc = nblk + wn * 64 + j * 16 + n;
      float bs = 0.f;
      if (bias0) bs += bias0[nc];
      if (bias1) bs += bias1[nc];
#pragma unroll
      for (int r = 0; r < 8; ++r) {
        const int m = mblk + wm * 32 + mi * 16 + r + (hi << 3);
        C[(size_t)m * N + nc] = acc[mi][j][r] + bs;
      }
    }
  }
}

// -----------------------------------------------------------------------------
// Sequential LSTM recurrence for one layer. One workgroup of 1024 threads
// (32 waves). h state (16x1024 bf16) in LDS; c state in per-lane registers.
// Each wave owns 2 hidden-column tiles and computes all 4 gates for them.
// -----------------------------------------------------------------------------
__global__ __launch_bounds__(1024) void lstm_rec_wmma(
    const float* __restrict__ xproj,   // (T*16, 4096) = x@Wih^T + b_ih + b_hh
    const bf16_t* __restrict__ Whh,    // (4096, 1024) bf16, row-major
    bf16_t* __restrict__ Hout,         // (T*16, 1024) bf16
    int Tn) {
  __shared__ bf16_t h_lds[16 * 1024];  // 32 KB
  const int lane = threadIdx.x & 31;
  const int wave = threadIdx.x >> 5;   // 0..31
  const int n = lane & 15, hi = lane >> 4;

  for (int i = threadIdx.x; i < 16 * 1024; i += blockDim.x) h_lds[i] = (bf16_t)0.f;
  __syncthreads();

  float cst[2][8];
#pragma unroll
  for (int tt = 0; tt < 2; ++tt)
#pragma unroll
    for (int r = 0; r < 8; ++r) cst[tt][r] = 0.f;

  for (int t = 0; t < Tn; ++t) {
    const float* xp = xproj + (size_t)t * 16 * 4096;

    v8f acc[2][4];
#pragma unroll
    for (int tt = 0; tt < 2; ++tt)
#pragma unroll
      for (int g = 0; g < 4; ++g)
#pragma unroll
        for (int r = 0; r < 8; ++r) acc[tt][g][r] = 0.f;

    for (int k0 = 0; k0 < 1024; k0 += 32) {
      v16bf a;  // A fragment: rows = batch (16), K window from h in LDS
      {
        const bf16_t* p = &h_lds[(lane & 15) * 1024 + k0];
#pragma unroll
        for (int e = 0; e < 16; ++e) a[e] = p[frag_k(e, hi)];
      }
#pragma unroll
      for (int tt = 0; tt < 2; ++tt) {
        const int j0 = (wave * 2 + tt) * 16;
#pragma unroll
        for (int g = 0; g < 4; ++g) {
          const bf16_t* pb = Whh + (size_t)(g * 1024 + j0 + (lane & 15)) * 1024 + k0;
          v16bf b;
#pragma unroll
          for (int e = 0; e < 16; ++e) b[e] = pb[frag_k(e, hi)];
          acc[tt][g] = __builtin_amdgcn_wmma_f32_16x16x32_bf16(
              false, a, false, b, (short)0, acc[tt][g], false, false);
        }
      }
    }
    __syncthreads();  // all waves done reading h_lds for this step

#pragma unroll
    for (int tt = 0; tt < 2; ++tt) {
      const int col = (wave * 2 + tt) * 16 + n;
#pragma unroll
      for (int r = 0; r < 8; ++r) {
        const int m = r + (hi << 3);            // batch row 0..15
        const float* row = xp + (size_t)m * 4096;
        float iv = acc[tt][0][r] + row[col];
        float fv = acc[tt][1][r] + row[1024 + col];
        float gv = acc[tt][2][r] + row[2048 + col];
        float ov = acc[tt][3][r] + row[3072 + col];
        iv = 1.f / (1.f + __expf(-iv));
        fv = 1.f / (1.f + __expf(-fv));
        gv = tanhf(gv);
        ov = 1.f / (1.f + __expf(-ov));
        float c = fv * cst[tt][r] + iv * gv;
        cst[tt][r] = c;
        float hv = ov * tanhf(c);
        bf16_t hb = (bf16_t)hv;
        h_lds[m * 1024 + col] = hb;
        Hout[((size_t)t * 16 + m) * 1024 + col] = hb;
      }
    }
    __syncthreads();  // new h visible before next step's reads
  }
}

__global__ void f32_to_bf16(const float* __restrict__ in, bf16_t* __restrict__ outp,
                            size_t nElem) {
  size_t i = (size_t)blockIdx.x * blockDim.x + threadIdx.x;
  size_t stride = (size_t)gridDim.x * blockDim.x;
  for (; i < nElem; i += stride) outp[i] = (bf16_t)in[i];
}

__global__ void embed_gather(const int* __restrict__ x, const float* __restrict__ emb,
                             bf16_t* __restrict__ X, int TBn, int En) {
  int idx = blockIdx.x * blockDim.x + threadIdx.x;
  if (idx >= TBn * En) return;
  int row = idx / En;
  int e = idx - row * En;
  X[idx] = (bf16_t)emb[(size_t)x[row] * En + e];
}

extern "C" void kernel_launch(void* const* d_in, const int* in_sizes, int n_in,
                              void* d_out, int out_size, void* d_ws, size_t ws_size,
                              hipStream_t stream) {
  (void)in_sizes; (void)n_in; (void)out_size; (void)ws_size;
  const int T = 256, B = 16, V = 32000, E = 1024, H = 1024, L = 2;
  const int TBr = T * B;       // 4096 rows
  const int G = 4 * H;         // 4096 gate columns

  const int*   x     = (const int*)d_in[0];
  const float* emb   = (const float*)d_in[1];
  const float* W_ih  = (const float*)d_in[2];
  const float* W_hh  = (const float*)d_in[3];
  const float* b_ih  = (const float*)d_in[4];
  const float* b_hh  = (const float*)d_in[5];
  const float* W_out = (const float*)d_in[6];
  const float* b_out = (const float*)d_in[7];
  float* out = (float*)d_out;

  char* ws = (char*)d_ws;
  size_t off = 0;
  auto carve = [&](size_t bytes) -> void* {
    size_t o = (off + 255) & ~(size_t)255;
    off = o + bytes;
    return (void*)(ws + o);
  };
  bf16_t* wih_b  = (bf16_t*)carve((size_t)L * G * E * sizeof(bf16_t));
  bf16_t* whh_b  = (bf16_t*)carve((size_t)L * G * H * sizeof(bf16_t));
  bf16_t* wout_b = (bf16_t*)carve((size_t)V * H * sizeof(bf16_t));
  bf16_t* Xa     = (bf16_t*)carve((size_t)TBr * E * sizeof(bf16_t));
  bf16_t* Xb     = (bf16_t*)carve((size_t)TBr * H * sizeof(bf16_t));
  float*  xproj  = (float*) carve((size_t)TBr * G * sizeof(float));

  // Weight conversion to bf16 (once per launch; deterministic).
  f32_to_bf16<<<2048, 256, 0, stream>>>(W_ih,  wih_b,  (size_t)L * G * E);
  f32_to_bf16<<<2048, 256, 0, stream>>>(W_hh,  whh_b,  (size_t)L * G * H);
  f32_to_bf16<<<4096, 256, 0, stream>>>(W_out, wout_b, (size_t)V * H);

  // Embedding gather -> bf16 activations.
  embed_gather<<<(TBr * E + 255) / 256, 256, 0, stream>>>(x, emb, Xa, TBr, E);

  bf16_t* cur = Xa;
  bf16_t* nxt = Xb;
  for (int l = 0; l < L; ++l) {
    // xproj = X @ W_ih[l]^T + b_ih[l] + b_hh[l]    (4096 x 4096, K=1024)
    gemm_bf16_wmma<<<dim3(TBr / 128, G / 256), 512, 0, stream>>>(
        cur, wih_b + (size_t)l * G * E,
        b_ih + (size_t)l * G, b_hh + (size_t)l * G,
        xproj, TBr, G, E);
    // Sequential recurrence over T on one WGP.
    lstm_rec_wmma<<<1, 1024, 0, stream>>>(
        xproj, whh_b + (size_t)l * G * H, nxt, T);
    bf16_t* tmp = cur; cur = nxt; nxt = tmp;
  }

  // logits = H @ W_out^T + b_out   (4096 x 32000, K=1024), fp32 out
  gemm_bf16_wmma<<<dim3(TBr / 128, V / 256), 512, 0, stream>>>(
      cur, wout_b, b_out, (const float*)nullptr, out, TBr, V, H);
}
